// SOAP_predictor_model_55362128445874
// MI455X (gfx1250) — matrix operand order repaired and checked by reading, not compiled, not run on hardware
//
#include <hip/hip_runtime.h>
#include <hip/hip_bf16.h>
#include <math.h>

typedef __attribute__((ext_vector_type(16))) _Float16 v16h;
typedef __attribute__((ext_vector_type(8)))  float    v8f;

#define N_ATOMS 64
#define E_ENV   10

__constant__ int c_LI[10] = {0,0,0,0,1,1,1,2,2,3};
__constant__ int c_HI[10] = {0,1,2,3,1,2,3,2,3,3};

// Hardware tanh (V_TANH_F32 is a gfx1250 TRANS32 op; co-executes with WMMA).
__device__ __forceinline__ float fast_tanh(float x) {
#if __has_builtin(__builtin_amdgcn_tanhf)
  return __builtin_amdgcn_tanhf(x);
#elif __has_builtin(__builtin_amdgcn_tanh_f32)
  return __builtin_amdgcn_tanh_f32(x);
#else
  return tanhf(x);
#endif
}

__device__ __forceinline__ float fast_sigmoid(float x) {
  return 1.0f / (1.0f + __expf(-x));
}

// ---------------------------------------------------------------------------
// K1: node-centre encodings. One block per atom n, 64 threads (one per m / d).
// ---------------------------------------------------------------------------
__global__ void __launch_bounds__(64) k_nce(
    const float* __restrict__ pos,
    const float* __restrict__ W0, const float* __restrict__ b0,
    const float* __restrict__ W1, const float* __restrict__ b1,
    const float* __restrict__ Wg, const float* __restrict__ Wv,
    float* __restrict__ nce)
{
  __shared__ float s_nce0m[64][16];
  __shared__ float s_nce0[16];
  __shared__ float s_e1[64][64];
  __shared__ float s_lg[64];
  __shared__ float s_w[64];
  __shared__ float s_mx, s_sum;
  const int n = blockIdx.x;
  const int t = threadIdx.x;   // 64 threads: t == m for rows, t == d for cols
  float pn[3], pm[3];
  for (int c = 0; c < 3; ++c) { pn[c] = pos[n*3+c]; pm[c] = pos[t*3+c]; }

  // nce0 = mean_m relu(pair6 @ W0 + b0)
  for (int k = 0; k < 16; ++k) {
    float v = b0[k];
    for (int c = 0; c < 3; ++c) v += pn[c]*W0[c*16+k] + pm[c]*W0[(3+c)*16+k];
    s_nce0m[t][k] = fmaxf(v, 0.f);
  }
  __syncthreads();
  if (t < 16) {
    float acc = 0.f;
    for (int m = 0; m < 64; ++m) acc += s_nce0m[m][t];
    s_nce0[t] = acc * (1.0f/64.0f);
  }
  __syncthreads();

  // e1[m=t,:] = relu(feat22 @ W1 + b1);  gate logit
  float lg = 0.f;
  for (int d = 0; d < 64; ++d) {
    float v = b1[d];
    for (int c = 0; c < 3; ++c) v += pn[c]*W1[c*64+d] + pm[c]*W1[(3+c)*64+d];
    for (int k = 0; k < 16; ++k) v += s_nce0[k]*W1[(6+k)*64+d];
    v = fmaxf(v, 0.f);
    s_e1[t][d] = v;
    lg += v * Wg[d];
  }
  s_lg[t] = lg;
  __syncthreads();
  if (t == 0) { float m = -1e30f; for (int i = 0; i < 64; ++i) m = fmaxf(m, s_lg[i]); s_mx = m; }
  __syncthreads();
  s_w[t] = __expf(s_lg[t] - s_mx);
  __syncthreads();
  if (t == 0) { float s = 0.f; for (int i = 0; i < 64; ++i) s += s_w[i]; s_sum = s; }
  __syncthreads();

  // nce[n][d=t] = sum_m w[m] * tanh(e1[m,:] @ Wv[:,d])
  float acc = 0.f;
  const float inv = 1.0f / s_sum;
  for (int m = 0; m < 64; ++m) {
    float dv = 0.f;
    for (int k = 0; k < 64; ++k) dv += s_e1[m][k] * Wv[k*64+t];
    acc += (s_w[m]*inv) * fast_tanh(dv);
  }
  nce[n*64+t] = acc;
}

// ---------------------------------------------------------------------------
// K2: h0, sequential GRU scan over 64 nodes, gated pool, final combine -> hG.
// Single block, 192 threads (one per GRU gate column).
// ---------------------------------------------------------------------------
__global__ void __launch_bounds__(192) k_graph(
    const float* __restrict__ nf,
    const float* __restrict__ Wa, const float* __restrict__ Wb,
    const float* __restrict__ Wx, const float* __restrict__ Wh, const float* __restrict__ b,
    const float* __restrict__ Wg, const float* __restrict__ Wv, const float* __restrict__ Wf,
    float* __restrict__ hG_out)
{
  __shared__ float s_h0[64][64];
  __shared__ float s_hs[64][64];
  __shared__ float s_h[64];
  __shared__ float s_gx[192], s_gh[192];
  __shared__ float s_lg[64], s_w[64];
  __shared__ float s_hg64[64];
  __shared__ float s_mx, s_sum;
  const int t = threadIdx.x;  // 192

  for (int idx = t; idx < 64*64; idx += 192) {
    int nn = idx >> 6, d = idx & 63;
    float v = 0.f;
    if (d < 16) { for (int j = 0; j < 13; ++j) v += nf[nn*13+j]*Wa[j*16+d]; }
    else        { for (int j = 0; j < 13; ++j) v += nf[nn*13+j]*Wb[j*48+(d-16)]; }
    s_h0[nn][d] = fmaxf(v, 0.f);
  }
  if (t < 64) s_h[t] = 0.f;
  __syncthreads();

  for (int step = 0; step < 64; ++step) {
    float gx = b[t], gh = 0.f;
    for (int k = 0; k < 64; ++k) { gx += s_h0[step][k]*Wx[k*192+t]; gh += s_h[k]*Wh[k*192+t]; }
    s_gx[t] = gx; s_gh[t] = gh;
    __syncthreads();
    float hnew = 0.f;
    if (t < 64) {
      float r  = fast_sigmoid(s_gx[t]      + s_gh[t]);
      float z  = fast_sigmoid(s_gx[64+t]   + s_gh[64+t]);
      float nn = fast_tanh(s_gx[128+t] + r*s_gh[128+t]);
      hnew = (1.f - z)*nn + z*s_h[t];
    }
    __syncthreads();
    if (t < 64) { s_h[t] = hnew; s_hs[step][t] = hnew; }
    __syncthreads();
  }

  if (t < 64) { float lg = 0.f; for (int k = 0; k < 64; ++k) lg += s_hs[t][k]*Wg[k]; s_lg[t] = lg; }
  __syncthreads();
  if (t == 0) { float m = -1e30f; for (int i = 0; i < 64; ++i) m = fmaxf(m, s_lg[i]); s_mx = m; }
  __syncthreads();
  if (t < 64) s_w[t] = __expf(s_lg[t] - s_mx);
  __syncthreads();
  if (t == 0) { float s = 0.f; for (int i = 0; i < 64; ++i) s += s_w[i]; s_sum = s; }
  __syncthreads();
  if (t < 64) {
    float acc = 0.f;
    const float inv = 1.0f / s_sum;
    for (int nn = 0; nn < 64; ++nn) {
      float dv = 0.f;
      for (int k = 0; k < 64; ++k) dv += s_hs[nn][k]*Wv[k*64+t];
      acc += (s_w[nn]*inv) * fast_tanh(dv);
    }
    s_hg64[t] = acc;
  }
  __syncthreads();
  if (t < 64) {
    float v = 0.f;
    for (int k = 0; k < 64; ++k) v += s_hg64[k]*Wf[k*64+t];
    for (int k = 0; k < 64; ++k) v += s_h0[63][k]*Wf[(64+k)*64+t];
    hG_out[t] = fast_tanh(v);
  }
}

// ---------------------------------------------------------------------------
// K3: gc[n] = GRU_step(hG, nce[n]). 64 blocks x 192 threads.
// ---------------------------------------------------------------------------
__global__ void __launch_bounds__(192) k_gc(
    const float* __restrict__ nce, const float* __restrict__ hG,
    const float* __restrict__ Wx, const float* __restrict__ Wh, const float* __restrict__ b,
    float* __restrict__ gc)
{
  __shared__ float s_gx[192], s_gh[192];
  const int n = blockIdx.x, t = threadIdx.x;
  float gx = b[t], gh = 0.f;
  for (int k = 0; k < 64; ++k) { gx += nce[n*64+k]*Wx[k*192+t]; gh += hG[k]*Wh[k*192+t]; }
  s_gx[t] = gx; s_gh[t] = gh;
  __syncthreads();
  if (t < 64) {
    float r  = fast_sigmoid(s_gx[t]      + s_gh[t]);
    float z  = fast_sigmoid(s_gx[64+t]   + s_gh[64+t]);
    float nn = fast_tanh(s_gx[128+t] + r*s_gh[128+t]);
    gc[n*64+t] = (1.f - z)*nn + z*hG[t];
  }
}

// ---------------------------------------------------------------------------
// K4: SOAP heads. One block per (atom a, env e), 256 threads (one per pair p),
// 4 heads looped inside. v0 = tanh(x0 @ Wv0) done with v_wmma_f32_16x16x32_f16;
// the gated aggregation is fused directly on the WMMA accumulator registers.
// Wv0 staged TRANSPOSED in LDS so both A and B fragments load as ds_load_b128.
// ---------------------------------------------------------------------------
__global__ void __launch_bounds__(256) k_soap(
    const float* __restrict__ pos, const float* __restrict__ gc,
    const float* __restrict__ hW0, const float* __restrict__ hb0,
    const float* __restrict__ hWg0, const float* __restrict__ hWv0,
    const float* __restrict__ hW1, const float* __restrict__ hb1,
    const float* __restrict__ hWg1, const float* __restrict__ hWv1,
    float* __restrict__ out)
{
  __shared__ __align__(16) unsigned char smem[60928];
  float*    rel6s = (float*)(smem + 0);        // [256][6]
  float*    gcv   = (float*)(smem + 6144);     // [64]
  float*    wsm   = (float*)(smem + 6400);     // [256] softmax weights
  float*    red   = (float*)(smem + 7424);     // [256] reduction scratch
  float*    agg0s = (float*)(smem + 8448);     // [64]
  // per-head phase A (x0 + WMMA)
  _Float16* x0h   = (_Float16*)(smem + 8704);  // [256][72] f16, K-padded stride
  _Float16* Wv0t  = (_Float16*)(smem + 45568); // [64 n][72 k] f16 (TRANSPOSED)
  float*    gcW0  = (float*)(smem + 54784);    // [64]
  float*    W0t   = (float*)(smem + 55040);    // [6][64]
  float*    Wg0s  = (float*)(smem + 56576);    // [64]
  float*    pagg  = (float*)(smem + 56832);    // [16][64]
  // layer1 (reuses x0h region — x0h dead by then)
  float*    v1s   = (float*)(smem + 8704);     // [256][16]
  float*    aggW1 = (float*)(smem + 25088);    // [16]
  float*    W1t   = (float*)(smem + 25152);    // [6][16]
  float*    Wg1s  = (float*)(smem + 25536);    // [16]
  float*    Wv1s  = (float*)(smem + 25600);    // [16][16]
  float*    pag1  = (float*)(smem + 26624);    // [16][16]

  const int bx = blockIdx.x;
  const int a = bx / E_ENV, e = bx % E_ENV;
  const int p = threadIdx.x;                 // 0..255 == pair index
  const int lane = p & 31, wid = p >> 5;
  const int lhalf = lane >> 4, lrow = lane & 15;

  { // rel6[p] = [posL[i]-pos[a], posH[j]-pos[a]]
    const int i = p >> 4, j = p & 15;
    const int L = c_LI[e], H = c_HI[e];
    for (int c = 0; c < 3; ++c) {
      float pa = pos[a*3+c];
      rel6s[p*6+c]   = pos[(L*16+i)*3+c] - pa;
      rel6s[p*6+3+c] = pos[(H*16+j)*3+c] - pa;
    }
  }
  if (p < 64) gcv[p] = gc[a*64+p];
  __syncthreads();

  const float r0 = rel6s[p*6+0], r1 = rel6s[p*6+1], r2 = rel6s[p*6+2],
              r3 = rel6s[p*6+3], r4 = rel6s[p*6+4], r5 = rel6s[p*6+5];

  for (int h = 0; h < 4; ++h) {
    // ---- head-constant vector gcW0 = gc @ W0[0:64] + b0; stage weights ----
    if (p < 64) {
      float v = hb0[h*64+p];
      for (int k = 0; k < 64; ++k) v += gcv[k]*hW0[(h*70+k)*64+p];
      gcW0[p] = v;
      Wg0s[p] = hWg0[h*64+p];
    }
    for (int idx = p; idx < 6*64; idx += 256)
      W0t[idx] = hW0[(h*70+64+(idx>>6))*64 + (idx&63)];
    for (int idx = p; idx < 64*64; idx += 256) {
      int k = idx >> 6, d = idx & 63;
      Wv0t[d*72+k] = (_Float16)hWv0[(h*64+k)*64+d];   // transposed stage
    }
    __syncthreads();

    // ---- x0[p,:] = relu(gcW0 + rel6 @ W0[64:70]) ; gate logit ----
    float lg = 0.f;
    _Float16* xrow = x0h + p*72;
    #pragma unroll 8
    for (int d = 0; d < 64; ++d) {
      float v = gcW0[d]
              + r0*W0t[d]       + r1*W0t[64+d]  + r2*W0t[128+d]
              + r3*W0t[192+d]   + r4*W0t[256+d] + r5*W0t[320+d];
      v = fmaxf(v, 0.f);
      xrow[d] = (_Float16)v;
      lg += v * Wg0s[d];
    }

    // ---- softmax over 256 pairs ----
    red[p] = lg; __syncthreads();
    for (int s = 128; s > 0; s >>= 1) { if (p < s) red[p] = fmaxf(red[p], red[p+s]); __syncthreads(); }
    float mx = red[0]; __syncthreads();
    float ex = __expf(lg - mx);
    red[p] = ex; __syncthreads();
    for (int s = 128; s > 0; s >>= 1) { if (p < s) red[p] += red[p+s]; __syncthreads(); }
    wsm[p] = ex / red[0];
    __syncthreads();

    // ---- WMMA: v0 = tanh(x0 @ Wv0), fused w0-weighted aggregation ----
    // M=256 (16 tiles), N=64 (4 tiles), K=64 (2 steps). 8 waves x 8 tiles.
    #pragma unroll
    for (int ni = 0; ni < 4; ++ni) {
      v16h b0f, b1f;   // B 32x16: lane = col N; K contiguous per half-wave
      {
        const _Float16* Br = Wv0t + (ni*16 + lrow)*72;
        const int kb0 = lhalf*16;
        #pragma unroll
        for (int i2 = 0; i2 < 16; ++i2) {
          b0f[i2] = Br[kb0+i2];
          b1f[i2] = Br[32+kb0+i2];
        }
      }
      float part = 0.f;
      #pragma unroll
      for (int tt = 0; tt < 2; ++tt) {
        const int mi = wid*2 + tt;
        v16h a0f, a1f; // A 16x32: lane = row M, K contiguous per half-wave
        const _Float16* Ar = x0h + (mi*16 + lrow)*72;
        const int kb = lhalf*16;
        #pragma unroll
        for (int i2 = 0; i2 < 8; ++i2) {
          a0f[i2]   = Ar[kb+i2];      a0f[i2+8] = Ar[kb+8+i2];
          a1f[i2]   = Ar[32+kb+i2];   a1f[i2+8] = Ar[32+kb+8+i2];
        }
        v8f cacc = {};
        cacc = __builtin_amdgcn_wmma_f32_16x16x32_f16(false, a0f, false, b0f, (short)0, cacc, false, false);
        cacc = __builtin_amdgcn_wmma_f32_16x16x32_f16(false, a1f, false, b1f, (short)0, cacc, false, false);
        const int mbase = mi*16 + lhalf*8;   // D: lane = col, VGPR r = row
        #pragma unroll
        for (int r = 0; r < 8; ++r) part += wsm[mbase+r] * fast_tanh(cacc[r]);
      }
      pagg[(wid*2+lhalf)*64 + ni*16 + lrow] = part;
    }
    __syncthreads();
    if (p < 64) {
      float acc = 0.f;
      for (int q = 0; q < 16; ++q) acc += pagg[q*64+p];
      agg0s[p] = acc;
    }
    __syncthreads();

    // ---- layer1 weight prep (agg0 is constant over p, so 70->16 collapses) ----
    if (p < 16) {
      float v = hb1[h*16+p];
      for (int k = 0; k < 64; ++k) v += agg0s[k]*hW1[(h*70+k)*16+p];
      aggW1[p] = v;
      Wg1s[p]  = hWg1[h*16+p];
    }
    if (p < 96)  W1t[p]  = hW1[(h*70+64+(p>>4))*16 + (p&15)];
    if (p < 16*16) Wv1s[p] = hWv1[h*256 + p];
    __syncthreads();

    // ---- x1, gate, softmax, v1, aggregation ----
    float x1r[16];
    float lg1 = 0.f;
    #pragma unroll
    for (int k = 0; k < 16; ++k) {
      float v = aggW1[k]
              + r0*W1t[k]    + r1*W1t[16+k] + r2*W1t[32+k]
              + r3*W1t[48+k] + r4*W1t[64+k] + r5*W1t[80+k];
      v = fmaxf(v, 0.f);
      x1r[k] = v;
      lg1 += v * Wg1s[k];
    }
    red[p] = lg1; __syncthreads();
    for (int s = 128; s > 0; s >>= 1) { if (p < s) red[p] = fmaxf(red[p], red[p+s]); __syncthreads(); }
    float mx1 = red[0]; __syncthreads();
    float ex1 = __expf(lg1 - mx1);
    red[p] = ex1; __syncthreads();
    for (int s = 128; s > 0; s >>= 1) { if (p < s) red[p] += red[p+s]; __syncthreads(); }
    float w1p = ex1 / red[0];
    __syncthreads();

    #pragma unroll
    for (int k2 = 0; k2 < 16; ++k2) {
      float dv = 0.f;
      #pragma unroll
      for (int k = 0; k < 16; ++k) dv += x1r[k]*Wv1s[k*16+k2];
      v1s[p*16+k2] = w1p * fast_tanh(dv);   // pre-multiplied by softmax weight
    }
    __syncthreads();
    {
      const int k2 = p & 15, cch = p >> 4;
      float part = 0.f;
      for (int q = 0; q < 16; ++q) part += v1s[(cch*16+q)*16 + k2];
      pag1[cch*16+k2] = part;
    }
    __syncthreads();
    if (p < 16) {
      float s = 0.f;
      for (int c2 = 0; c2 < 16; ++c2) s += pag1[c2*16+p];
      out[((h*64 + a)*10 + e)*16 + p] = s;   // SOAP [4,N,E,16]
    }
    __syncthreads();
  }
}

// ---------------------------------------------------------------------------
extern "C" void kernel_launch(void* const* d_in, const int* in_sizes, int n_in,
                              void* d_out, int out_size, void* d_ws, size_t ws_size,
                              hipStream_t stream)
{
  (void)in_sizes; (void)n_in; (void)out_size; (void)ws_size;
  const float* positions = (const float*)d_in[0];
  const float* node_feats= (const float*)d_in[1];
  const float* nce_W0 = (const float*)d_in[2];
  const float* nce_b0 = (const float*)d_in[3];
  const float* nce_W1 = (const float*)d_in[4];
  const float* nce_b1 = (const float*)d_in[5];
  const float* nce_Wg = (const float*)d_in[6];
  const float* nce_Wv = (const float*)d_in[7];
  const float* h0_Wa  = (const float*)d_in[8];
  const float* h0_Wb  = (const float*)d_in[9];
  const float* gru_Wx = (const float*)d_in[10];
  const float* gru_Wh = (const float*)d_in[11];
  const float* gru_b  = (const float*)d_in[12];
  const float* hG_Wg  = (const float*)d_in[13];
  const float* hG_Wv  = (const float*)d_in[14];
  const float* final_Wf = (const float*)d_in[15];
  const float* gc_Wx  = (const float*)d_in[16];
  const float* gc_Wh  = (const float*)d_in[17];
  const float* gc_b   = (const float*)d_in[18];
  const float* hW0    = (const float*)d_in[19];
  const float* hb0    = (const float*)d_in[20];
  const float* hWg0   = (const float*)d_in[21];
  const float* hWv0   = (const float*)d_in[22];
  const float* hW1    = (const float*)d_in[23];
  const float* hb1    = (const float*)d_in[24];
  const float* hWg1   = (const float*)d_in[25];
  const float* hWv1   = (const float*)d_in[26];
  float* out = (float*)d_out;
  float* ws  = (float*)d_ws;
  float* nce = ws;               // 64*64
  float* hG  = ws + 4096;        // 64
  float* gc  = ws + 4096 + 64;   // 64*64

  k_nce  <<<64,  64,  0, stream>>>(positions, nce_W0, nce_b0, nce_W1, nce_b1, nce_Wg, nce_Wv, nce);
  k_graph<<<1,   192, 0, stream>>>(node_feats, h0_Wa, h0_Wb, gru_Wx, gru_Wh, gru_b, hG_Wg, hG_Wv, final_Wf, hG);
  k_gc   <<<64,  192, 0, stream>>>(nce, hG, gc_Wx, gc_Wh, gc_b, gc);
  k_soap <<<640, 256, 0, stream>>>(positions, gc, hW0, hb0, hWg0, hWv0, hW1, hb1, hWg1, hWv1, out);
}
